// SingleStreamBlock_86629490360824
// MI455X (gfx1250) — matrix operand order repaired and checked by reading, not compiled, not run on hardware
//
#include <hip/hip_runtime.h>
#include <hip/hip_bf16.h>

// ---------------------------------------------------------------------------
// CDNA5 (gfx1250, wave32) implementation of the MMDiT SingleStreamBlock.
// All GEMMs + attention matmuls run on v_wmma_f32_16x16x32_f16.
// LDS tile staging uses the Tensor Data Mover (tensor_load_to_lds +
// s_wait_tensorcnt) when the builtin is available, else plain loads.
// ---------------------------------------------------------------------------

typedef __attribute__((ext_vector_type(16))) _Float16 v16h;
typedef __attribute__((ext_vector_type(8)))  _Float16 v8h;
typedef __attribute__((ext_vector_type(8)))  float    v8f;
typedef unsigned int u32x4 __attribute__((ext_vector_type(4)));
typedef int          i32x8 __attribute__((ext_vector_type(8)));
typedef int          i32x4 __attribute__((ext_vector_type(4)));

#define DEV __device__ __forceinline__

#if defined(__has_builtin)
#if __has_builtin(__builtin_amdgcn_tensor_load_to_lds) && \
    __has_builtin(__builtin_amdgcn_s_wait_tensorcnt)
#define HAVE_TDM 1
#endif
#endif

constexpr int BATCH  = 2;
constexpr int LTXT   = 256;
constexpr int NIMG   = 2048;
constexpr int SEQ    = LTXT + NIMG;    // 2304
constexpr int NT     = BATCH * SEQ;    // 4608 tokens total
constexpr int DMODEL = 1536;
constexpr int NH     = 12;
constexpr int DH     = 128;
constexpr int QKVD   = 3 * NH * DH;    // 4608
constexpr int MLPH   = 6144;
constexpr float RMS_EPS  = 1e-6f;
constexpr float QK_SCALE = 0.08838834764831845f;  // DH^-0.5

// ------------------------------- WMMA helpers ------------------------------

DEV v8f wmma16(v16h a, v16h b, v8f c) {
  // D = A(16x32 f16) * B(32x16 f16) + C(16x16 f32)
  return __builtin_amdgcn_wmma_f32_16x16x32_f16(
      /*neg_a=*/false, a, /*neg_b=*/false, b,
      /*c_mod=*/(short)0, c, /*reuse_a=*/false, /*reuse_b=*/false);
}

// A-matrix fragment (16x32, f16) from a row-major [rows x ld] buffer.
// ISA 7.12.2: lane L holds row M=L&15; lanes<16 carry K {0..7,16..23},
// lanes>=16 carry K {8..15,24..31} -> two contiguous 8-half loads per lane.
DEV v16h frag_a(const _Float16* __restrict__ p0, int ld, int row0, int k0) {
  const int lane = threadIdx.x & 31;
  const int sel  = (lane < 16) ? 0 : 8;
  const _Float16* p = p0 + (size_t)(row0 + (lane & 15)) * ld + k0 + sel;
  v8h lo = *(const v8h*)(p);
  v8h hi = *(const v8h*)(p + 16);
  v16h a;
#pragma unroll
  for (int i = 0; i < 8; ++i) { a[i] = lo[i]; a[i + 8] = hi[i]; }
  return a;
}

// B-matrix fragment (32x16, f16) where B(k,n) = M[n*ld + k] for a row-major
// matrix M (i.e. B = M^T). Lane L holds column N=L&15; lanes<16 carry
// K {0..7,16..23}, lanes>=16 carry K {8..15,24..31} -> two 8-half loads.
DEV v16h frag_b(const _Float16* __restrict__ p0, int ld, int col0, int k0) {
  const int lane = threadIdx.x & 31;
  const int koff = k0 + ((lane < 16) ? 0 : 8);
  const _Float16* p = p0 + (size_t)(col0 + (lane & 15)) * ld + koff;
  v8h lo = *(const v8h*)(p);
  v8h hi = *(const v8h*)(p + 16);
  v16h b;
#pragma unroll
  for (int i = 0; i < 8; ++i) { b[i] = lo[i]; b[i + 8] = hi[i]; }
  return b;
}

// ------------------------------- TDM helpers -------------------------------
#if defined(HAVE_TDM)
DEV unsigned int lds_off(const void* p) {
  // generic -> LDS address space -> 32-bit byte offset within WG LDS
  return (unsigned int)(__UINTPTR_TYPE__)(
      __attribute__((address_space(3))) const void*)p;
}

// 2-D tile load, f16 elements. tile_d0 = contiguous halves per row,
// tile_d1 = rows, stride = row pitch in halves. Pad codes per ISA 8.4:
// pad after every 2^(pic+1) DWORDs, pad amount (pac+1) DWORDs.
DEV void tdm_load_2d(unsigned int lds_byte_off, const void* gaddr,
                     unsigned int tile_d0, unsigned int tile_d1,
                     unsigned long long stride_halves,
                     unsigned int pic, unsigned int pac) {
  unsigned long long ga = (unsigned long long)(__UINTPTR_TYPE__)gaddr;
  u32x4 g0;
  g0[0] = 1u;                                   // count=1, user descriptor
  g0[1] = lds_byte_off;                         // lds_addr
  g0[2] = (unsigned int)ga;                     // global_addr[31:0]
  g0[3] = (unsigned int)((ga >> 32) & 0x1FFFFFFu) | (2u << 30);  // type=2
  i32x8 g1;
  g1[0] = (int)((1u << 16)        // data_size = 2 bytes
              | (1u << 20)        // pad_enable
              | (pic << 22)       // pad_interval code
              | (pac << 25));     // pad_amount code
  g1[1] = (int)((tile_d0 & 0xFFFFu) << 16);             // tensor_dim0 lo16
  g1[2] = (int)((tile_d0 >> 16) | ((tile_d1 & 0xFFFFu) << 16));  // d0 hi/d1 lo
  g1[3] = (int)((tile_d1 >> 16) | ((tile_d0 & 0xFFFFu) << 16));  // tile_dim0
  g1[4] = (int)(tile_d1 & 0xFFFFu);                     // tile_dim1
  g1[5] = (int)(unsigned int)(stride_halves & 0xFFFFFFFFu);      // stride lo
  g1[6] = (int)(unsigned int)((stride_halves >> 32) & 0xFFFFu);  // stride hi
  g1[7] = 0;
  i32x4 z4 = {0, 0, 0, 0};
  i32x8 z8 = {0, 0, 0, 0, 0, 0, 0, 0};
  // 6-arg form (clang-23 / therock toolchain): groups 0..3 + extra + cpol
  __builtin_amdgcn_tensor_load_to_lds(g0, g1, z4, z4, z8, 0);
}
#endif

// ------------------------------ small helpers ------------------------------

DEV float gelu_tanh(float x) {
  float x3 = x * x * x;
  float t  = tanhf(0.7978845608028654f * (x + 0.044715f * x3));
  return 0.5f * x * (1.0f + t);
}

DEV float block_reduce_sum_256(float v) {
  __shared__ float red[8];
  __shared__ float tot;
  const int lane = threadIdx.x & 31;
  const int wid  = threadIdx.x >> 5;
#pragma unroll
  for (int off = 16; off; off >>= 1) v += __shfl_xor(v, off, 32);
  if (lane == 0) red[wid] = v;
  __syncthreads();
  if (threadIdx.x == 0) {
    float t = 0.f;
#pragma unroll
    for (int i = 0; i < 8; ++i) t += red[i];
    tot = t;
  }
  __syncthreads();
  return tot;
}

// --------------------------- elementwise kernels ---------------------------

__global__ __launch_bounds__(256) void ssb_cvt_f32_to_f16(
    const float* __restrict__ src, _Float16* __restrict__ dst, size_t n) {
  size_t i = (size_t)blockIdx.x * 256 + threadIdx.x;
  size_t stride = (size_t)gridDim.x * 256;
  for (; i < n; i += stride) dst[i] = (_Float16)src[i];
}

// modout[b, j] = silu(vec[b]) . w_mod[j] + b_mod[j]   (B x 4608 GEMV)
__global__ __launch_bounds__(256) void ssb_mod_kernel(
    const float* __restrict__ vec, const float* __restrict__ w_mod,
    const float* __restrict__ b_mod, float* __restrict__ out) {
  const int j   = blockIdx.x;            // 0 .. B*3D-1
  const int b   = j / (3 * DMODEL);
  const int row = j % (3 * DMODEL);
  float s = 0.f;
  for (int d = threadIdx.x; d < DMODEL; d += 256) {
    float v  = vec[b * DMODEL + d];
    float sv = v / (1.0f + __expf(-v));   // silu
    s += sv * w_mod[(size_t)row * DMODEL + d];
  }
  float tot = block_reduce_sum_256(s);
  if (threadIdx.x == 0) out[j] = tot + b_mod[row];
}

// concat(txt,img) -> x (fp32 residual) ; RMSNorm*w*(1+scale)+shift -> xn (f16)
__global__ __launch_bounds__(256) void ssb_norm1_kernel(
    const float* __restrict__ txt, const float* __restrict__ img,
    const float* __restrict__ modv, const float* __restrict__ w_norm1,
    float* __restrict__ x, _Float16* __restrict__ xn) {
  const int rowid = blockIdx.x;          // 0 .. NT-1
  const int b = rowid / SEQ;
  const int n = rowid % SEQ;
  const float* src = (n < LTXT)
      ? (txt + ((size_t)b * LTXT + n) * DMODEL)
      : (img + ((size_t)b * NIMG + (n - LTXT)) * DMODEL);
  float vals[6];
  float ss = 0.f;
#pragma unroll
  for (int i = 0; i < 6; ++i) {
    int d = threadIdx.x + i * 256;
    vals[i] = src[d];
    ss += vals[i] * vals[i];
  }
  float tot = block_reduce_sum_256(ss);
  float inv = rsqrtf(tot / (float)DMODEL + RMS_EPS);
  const float* shift = modv + b * 3 * DMODEL;
  const float* scale = shift + DMODEL;
#pragma unroll
  for (int i = 0; i < 6; ++i) {
    int d = threadIdx.x + i * 256;
    size_t o = (size_t)rowid * DMODEL + d;
    x[o] = vals[i];
    float xv = vals[i] * inv * w_norm1[d] * (1.0f + scale[d]) + shift[d];
    xn[o] = (_Float16)xv;
  }
}

// qkv (B,N,3,H,DH) f16  -> rope(q),rope(k) as (B,H,N,DH) f16, v as (B,H,DH,N)
__global__ __launch_bounds__(256) void ssb_rope_split_kernel(
    const _Float16* __restrict__ qkv, const float* __restrict__ img_rope,
    _Float16* __restrict__ q, _Float16* __restrict__ k,
    _Float16* __restrict__ vt) {
  const size_t idx = (size_t)blockIdx.x * 256 + threadIdx.x;
  const size_t total = (size_t)BATCH * SEQ * NH * (DH / 2);
  if (idx >= total) return;
  const int d2 = idx % (DH / 2);
  const int h  = (idx / (DH / 2)) % NH;
  const int n  = (idx / ((DH / 2) * NH)) % SEQ;
  const int b  = idx / ((size_t)(DH / 2) * NH * SEQ);
  const int d0 = 2 * d2;

  const _Float16* base = qkv + (size_t)(b * SEQ + n) * QKVD + h * DH + d0;
  float q0 = (float)base[0];
  float q1 = (float)base[1];
  float k0 = (float)base[NH * DH + 0];
  float k1 = (float)base[NH * DH + 1];
  float v0 = (float)base[2 * NH * DH + 0];
  float v1 = (float)base[2 * NH * DH + 1];

  float cs = 0.f, sn = 0.f;   // txt positions get zero rope (per reference)
  if (n >= LTXT) {
    const float* rp = img_rope + ((size_t)b * NIMG + (n - LTXT)) * DH + d0;
    cs = rp[0];
    sn = rp[1];
  }
  float qo0 = q0 * cs - q1 * sn, qo1 = q1 * cs + q0 * sn;
  float ko0 = k0 * cs - k1 * sn, ko1 = k1 * cs + k0 * sn;

  const size_t bh = (size_t)b * NH + h;
  _Float16* qdst = q + (bh * SEQ + n) * DH + d0;
  _Float16* kdst = k + (bh * SEQ + n) * DH + d0;
  qdst[0] = (_Float16)qo0; qdst[1] = (_Float16)qo1;
  kdst[0] = (_Float16)ko0; kdst[1] = (_Float16)ko1;
  // V transposed: vt[b][h][d][n]
  vt[(bh * DH + d0 + 0) * SEQ + n] = (_Float16)v0;
  vt[(bh * DH + d0 + 1) * SEQ + n] = (_Float16)v1;
}

// x += gate * proj ; then RMSNorm(x)*w_norm2 -> xn2 (f16)
__global__ __launch_bounds__(256) void ssb_resid_norm2_kernel(
    float* __restrict__ x, const float* __restrict__ proj,
    const float* __restrict__ modv, const float* __restrict__ w_norm2,
    _Float16* __restrict__ xn2) {
  const int rowid = blockIdx.x;
  const int b = rowid / SEQ;
  const float* gate = modv + b * 3 * DMODEL + 2 * DMODEL;
  float vals[6];
  float ss = 0.f;
#pragma unroll
  for (int i = 0; i < 6; ++i) {
    int d = threadIdx.x + i * 256;
    size_t o = (size_t)rowid * DMODEL + d;
    float xv = x[o] + gate[d] * proj[o];
    vals[i] = xv;
    x[o] = xv;
    ss += xv * xv;
  }
  float tot = block_reduce_sum_256(ss);
  float inv = rsqrtf(tot / (float)DMODEL + RMS_EPS);
#pragma unroll
  for (int i = 0; i < 6; ++i) {
    int d = threadIdx.x + i * 256;
    size_t o = (size_t)rowid * DMODEL + d;
    xn2[o] = (_Float16)(vals[i] * inv * w_norm2[d]);
  }
}

// out = x + fc2out + b_fc2 split into (txt_out, img_out) flat in d_out
__global__ __launch_bounds__(256) void ssb_final_kernel(
    const float* __restrict__ x, const float* __restrict__ fc2out,
    const float* __restrict__ b_fc2, float* __restrict__ out) {
  const size_t idx = (size_t)blockIdx.x * 256 + threadIdx.x;
  const size_t total = (size_t)NT * DMODEL;
  if (idx >= total) return;
  const int d = idx % DMODEL;
  const int n = (idx / DMODEL) % SEQ;
  const int b = idx / ((size_t)DMODEL * SEQ);
  float v = x[idx] + fc2out[idx] + b_fc2[d];
  if (n < LTXT) {
    out[((size_t)b * LTXT + n) * DMODEL + d] = v;
  } else {
    const size_t txt_elems = (size_t)BATCH * LTXT * DMODEL;
    out[txt_elems + ((size_t)b * NIMG + (n - LTXT)) * DMODEL + d] = v;
  }
}

// ------------------------------- WMMA GEMM --------------------------------
// C[M,N] = A[M,K](f16,row-major) * W[N,K](f16,row-major)^T  (+ epilogue)
// Block: 256 threads = 8 waves; block tile 128x64; wave tile 32x32.
// Tiles staged into LDS by the Tensor Data Mover when available.
// EPI: 0 = store fp32, 1 = store f16, 2 = bias + tanh-GELU -> f16
template <int EPI>
__global__ __launch_bounds__(256) void ssb_gemm_wmma_kernel(
    const _Float16* __restrict__ A, const _Float16* __restrict__ W,
    const float* __restrict__ bias, void* __restrict__ Cout,
    int M, int N, int K) {
  constexpr int LDT = 40;  // padded LDS stride (halves)
  __shared__ _Float16 As[128 * LDT];
  __shared__ _Float16 Bs[64 * LDT];

  const int m0  = blockIdx.y * 128;
  const int n0  = blockIdx.x * 64;
  const int wid = threadIdx.x >> 5;
  const int wm  = wid & 3;   // 0..3 -> 32-row slab
  const int wn  = wid >> 2;  // 0..1 -> 32-col slab

  v8f acc00 = {}, acc01 = {}, acc10 = {}, acc11 = {};

  for (int k0 = 0; k0 < K; k0 += 32) {
#if defined(HAVE_TDM)
    if (wid == 0) {
      // pad codes (3,3): +4 DWORDs after each 16 DWORDs -> ld = 40 halves
      tdm_load_2d(lds_off(As), A + (size_t)m0 * K + k0, 32u, 128u,
                  (unsigned long long)K, 3u, 3u);
      tdm_load_2d(lds_off(Bs), W + (size_t)n0 * K + k0, 32u, 64u,
                  (unsigned long long)K, 3u, 3u);
      __builtin_amdgcn_s_wait_tensorcnt(0);
    }
#else
    {  // stage A tile: 128 rows x 32 halves, 1 row-segment of 16 per thread
      const int row = threadIdx.x >> 1;
      const int seg = (threadIdx.x & 1) << 4;
      const _Float16* src = A + (size_t)(m0 + row) * K + k0 + seg;
      _Float16* dst = As + row * LDT + seg;
      *(v8h*)(dst)     = *(const v8h*)(src);
      *(v8h*)(dst + 8) = *(const v8h*)(src + 8);
    }
    {  // stage W tile: 64 rows x 32 halves, 8 halves per thread
      const int row = threadIdx.x >> 2;
      const int seg = (threadIdx.x & 3) << 3;
      const _Float16* src = W + (size_t)(n0 + row) * K + k0 + seg;
      *(v8h*)(Bs + row * LDT + seg) = *(const v8h*)(src);
    }
#endif
    __syncthreads();
    const v16h a0 = frag_a(As, LDT, wm * 32 + 0, 0);
    const v16h a1 = frag_a(As, LDT, wm * 32 + 16, 0);
    const v16h b0 = frag_b(Bs, LDT, wn * 32 + 0, 0);
    const v16h b1 = frag_b(Bs, LDT, wn * 32 + 16, 0);
    acc00 = wmma16(a0, b0, acc00);
    acc01 = wmma16(a0, b1, acc01);
    acc10 = wmma16(a1, b0, acc10);
    acc11 = wmma16(a1, b1, acc11);
    __syncthreads();
  }

  // store: C-layout -> VGPR r holds row (r | lanehi*8), col = lane&15
  const int lane = threadIdx.x & 31;
  const int cn   = lane & 15;
  const int rb   = (lane < 16) ? 0 : 8;
  const v8f* accs[4] = {&acc00, &acc01, &acc10, &acc11};
#pragma unroll
  for (int mi = 0; mi < 2; ++mi) {
#pragma unroll
    for (int ni = 0; ni < 2; ++ni) {
      const v8f& c = *accs[mi * 2 + ni];
#pragma unroll
      for (int r = 0; r < 8; ++r) {
        const size_t row = (size_t)m0 + wm * 32 + mi * 16 + rb + r;
        const size_t col = (size_t)n0 + wn * 32 + ni * 16 + cn;
        float v = c[r];
        if constexpr (EPI == 0) {
          ((float*)Cout)[row * N + col] = v;
        } else if constexpr (EPI == 1) {
          ((_Float16*)Cout)[row * N + col] = (_Float16)v;
        } else {
          v += bias[col];
          ((_Float16*)Cout)[row * N + col] = (_Float16)gelu_tanh(v);
        }
      }
    }
  }
}

// --------------------------- flash attention -------------------------------
// Block = 8 waves, all sharing one (b,h) (8 divides 144 q-tiles/head).
// K/V 32-key tiles are staged once per block into LDS (TDM when available),
// each wave runs online softmax for its own 16-query tile.
// q,k: (B,H,SEQ,DH) f16 ; vt: (B,H,DH,SEQ) f16 ; out: (B,SEQ,H*DH) f16
__global__ __launch_bounds__(256) void ssb_attn_kernel(
    const _Float16* __restrict__ q16, const _Float16* __restrict__ k16,
    const _Float16* __restrict__ vt16, _Float16* __restrict__ attnO) {
  constexpr int QTILES = SEQ / 16;          // 144
  constexpr int LDK = DH + 8;               // 136: K-tile LDS stride
  constexpr int LDV = 40;                   // V-tile LDS stride
  __shared__ _Float16 Ks[32 * LDK];         // 32 keys x 128 dims
  __shared__ _Float16 Vs[DH * LDV];         // 128 dims x 32 keys
  __shared__ _Float16 Pbuf[8][16 * 40];     // per-wave P bounce buffer

  const int wid  = threadIdx.x >> 5;
  const int lane = threadIdx.x & 31;
  const int task = blockIdx.x * 8 + wid;    // grid sized exactly
  const int qt   = task % QTILES;
  const int bh   = task / QTILES;           // identical for all 8 waves
  const int b    = bh / NH;
  const int h    = bh % NH;

  const _Float16* qp = q16 + ((size_t)bh * SEQ + qt * 16) * DH;
  const _Float16* kp = k16 + (size_t)bh * SEQ * DH;
  const _Float16* vp = vt16 + (size_t)bh * DH * SEQ;
  _Float16* Pl = Pbuf[wid];

  // Q fragments: 16 rows x DH, four K=32 chunks (held in registers)
  v16h qa[4];
#pragma unroll
  for (int c = 0; c < 4; ++c) qa[c] = frag_a(qp, DH, 0, c * 32);

  v8f   o[8] = {};                          // 16 x 128 f32 accumulator
  float m[8], l[8];
#pragma unroll
  for (int r = 0; r < 8; ++r) { m[r] = -1e30f; l[r] = 0.f; }

  for (int kt = 0; kt < SEQ; kt += 32) {
    // ---- cooperative staging of K (32 x 128) and V^T (128 x 32) tiles ----
#if defined(HAVE_TDM)
    if (wid == 0) {
      // K rows: 128 halves (64 DW) -> pad codes (5,3) => ld = 136 halves
      tdm_load_2d(lds_off(Ks), kp + (size_t)kt * DH, (unsigned)DH, 32u,
                  (unsigned long long)DH, 5u, 3u);
      // V rows: 32 halves (16 DW) -> pad codes (3,3) => ld = 40 halves
      tdm_load_2d(lds_off(Vs), vp + kt, 32u, (unsigned)DH,
                  (unsigned long long)SEQ, 3u, 3u);
      __builtin_amdgcn_s_wait_tensorcnt(0);
    }
#else
    {  // K tile: 32 rows x 128 halves; 16 halves per thread
      const int r = threadIdx.x >> 3;
      const int s = (threadIdx.x & 7) * 16;
      const _Float16* src = kp + (size_t)(kt + r) * DH + s;
      _Float16* dst = Ks + r * LDK + s;
      *(v8h*)(dst)     = *(const v8h*)(src);
      *(v8h*)(dst + 8) = *(const v8h*)(src + 8);
    }
    {  // V tile: 128 rows x 32 halves; 16 halves per thread
      const int r = threadIdx.x >> 1;
      const int s = (threadIdx.x & 1) * 16;
      const _Float16* src = vp + (size_t)r * SEQ + kt + s;
      _Float16* dst = Vs + r * LDV + s;
      *(v8h*)(dst)     = *(const v8h*)(src);
      *(v8h*)(dst + 8) = *(const v8h*)(src + 8);
    }
#endif
    __syncthreads();

    // ---- S = Q * K^T for 32 keys (two 16x16 tiles) ----
    v8f s0 = {}, s1 = {};
#pragma unroll
    for (int c = 0; c < 4; ++c) {
      const v16h kb0 = frag_b(Ks, LDK, 0, c * 32);
      const v16h kb1 = frag_b(Ks, LDK, 16, c * 32);
      s0 = wmma16(qa[c], kb0, s0);
      s1 = wmma16(qa[c], kb1, s1);
    }

    // ---- online softmax (rows split across lane halves; width-16 reduce) --
    const int row = ((lane < 16) ? 0 : 8);
#pragma unroll
    for (int r = 0; r < 8; ++r) {
      float e0 = s0[r] * QK_SCALE;
      float e1 = s1[r] * QK_SCALE;
      float mx = fmaxf(e0, e1);
#pragma unroll
      for (int off = 8; off; off >>= 1) mx = fmaxf(mx, __shfl_xor(mx, off, 16));
      float mnew = fmaxf(m[r], mx);
      float corr = __expf(m[r] - mnew);
      float p0 = __expf(e0 - mnew);
      float p1 = __expf(e1 - mnew);
      float rs = p0 + p1;
#pragma unroll
      for (int off = 8; off; off >>= 1) rs += __shfl_xor(rs, off, 16);
      l[r] = l[r] * corr + rs;
      m[r] = mnew;
#pragma unroll
      for (int f = 0; f < 8; ++f) o[f][r] *= corr;
      // P tile (16x32) into per-wave LDS for re-layout into an A fragment
      Pl[(row + r) * 40 + (lane & 15)]      = (_Float16)p0;
      Pl[(row + r) * 40 + 16 + (lane & 15)] = (_Float16)p1;
    }
    asm volatile("s_wait_dscnt 0" ::: "memory");   // CDNA5 split LDS counter
    __builtin_amdgcn_wave_barrier();

    const v16h pa = frag_a(Pl, 40, 0, 0);
    // ---- O += P * V from the shared V tile ----
#pragma unroll
    for (int f = 0; f < 8; ++f) {
      const v16h vb = frag_b(Vs, LDV, f * 16, 0);
      o[f] = wmma16(pa, vb, o[f]);
    }
    __syncthreads();
  }

  // normalize and store to (B, SEQ, H*DH)
  const int rb = (lane < 16) ? 0 : 8;
#pragma unroll
  for (int f = 0; f < 8; ++f) {
#pragma unroll
    for (int r = 0; r < 8; ++r) {
      const int n = qt * 16 + rb + r;
      const int d = f * 16 + (lane & 15);
      float v = o[f][r] / l[r];
      attnO[((size_t)b * SEQ + n) * (NH * DH) + h * DH + d] = (_Float16)v;
    }
  }
}

// ------------------------------- launcher ----------------------------------

extern "C" void kernel_launch(void* const* d_in, const int* in_sizes, int n_in,
                              void* d_out, int out_size, void* d_ws,
                              size_t ws_size, hipStream_t stream) {
  (void)in_sizes; (void)n_in; (void)out_size; (void)ws_size;

  const float* txt      = (const float*)d_in[0];
  const float* img      = (const float*)d_in[1];
  const float* vec      = (const float*)d_in[2];
  /* d_in[3] = txt_rope : ignored by the reference */
  const float* img_rope = (const float*)d_in[4];
  const float* w_norm1  = (const float*)d_in[5];
  const float* w_mod    = (const float*)d_in[6];
  const float* b_mod    = (const float*)d_in[7];
  const float* w_qkv    = (const float*)d_in[8];
  const float* w_out    = (const float*)d_in[9];
  const float* w_norm2  = (const float*)d_in[10];
  const float* w_fc1    = (const float*)d_in[11];
  const float* b_fc1    = (const float*)d_in[12];
  const float* w_fc2    = (const float*)d_in[13];
  const float* b_fc2    = (const float*)d_in[14];
  float* out = (float*)d_out;

  // workspace sub-allocation
  char* ws = (char*)d_ws;
  size_t off = 0;
  auto take = [&](size_t bytes) -> void* {
    void* p = ws + off;
    off += (bytes + 255) & ~(size_t)255;
    return p;
  };
  float*    x      = (float*)   take((size_t)NT * DMODEL * 4);
  _Float16* xn16   = (_Float16*)take((size_t)NT * DMODEL * 2);
  _Float16* wqkv16 = (_Float16*)take((size_t)QKVD * DMODEL * 2);
  _Float16* wout16 = (_Float16*)take((size_t)DMODEL * DMODEL * 2);
  _Float16* wfc116 = (_Float16*)take((size_t)MLPH * DMODEL * 2);
  _Float16* wfc216 = (_Float16*)take((size_t)DMODEL * MLPH * 2);
  float*    modout = (float*)   take((size_t)BATCH * 3 * DMODEL * 4);
  _Float16* attnO  = (_Float16*)take((size_t)NT * DMODEL * 2);
  float*    c32    = (float*)   take((size_t)NT * DMODEL * 4);
  _Float16* qkv16  = (_Float16*)take((size_t)NT * QKVD * 2);
  _Float16* q16    = (_Float16*)take((size_t)NT * DMODEL * 2);
  _Float16* k16    = (_Float16*)take((size_t)NT * DMODEL * 2);
  _Float16* vt16   = (_Float16*)take((size_t)NT * DMODEL * 2);
  // MLP hidden (f16) aliases qkv16/q16 region (dead after attention)
  _Float16* h16 = qkv16;

  const dim3 blk(256);

  // 1) weight conversions fp32 -> f16
  {
    auto cvt = [&](const float* s, _Float16* d, size_t n) {
      ssb_cvt_f32_to_f16<<<dim3((unsigned)((n + 255) / 256)), blk, 0, stream>>>(s, d, n);
    };
    cvt(w_qkv, wqkv16, (size_t)QKVD * DMODEL);
    cvt(w_out, wout16, (size_t)DMODEL * DMODEL);
    cvt(w_fc1, wfc116, (size_t)MLPH * DMODEL);
    cvt(w_fc2, wfc216, (size_t)DMODEL * MLPH);
  }

  // 2) AdaLN modulation: silu(vec) @ w_mod^T + b_mod
  ssb_mod_kernel<<<dim3(BATCH * 3 * DMODEL), blk, 0, stream>>>(vec, w_mod, b_mod, modout);

  // 3) concat + RMSNorm1 + (1+scale)/shift modulation
  ssb_norm1_kernel<<<dim3(NT), blk, 0, stream>>>(txt, img, modout, w_norm1, x, xn16);

  // 4) QKV GEMM: [4608,1536] x [4608,1536]^T -> f16
  ssb_gemm_wmma_kernel<1><<<dim3(QKVD / 64, NT / 128), blk, 0, stream>>>(
      xn16, wqkv16, nullptr, qkv16, NT, QKVD, DMODEL);

  // 5) RoPE (txt rope = zeros) + split into q,k,(V^T)
  {
    const size_t total = (size_t)BATCH * SEQ * NH * (DH / 2);
    ssb_rope_split_kernel<<<dim3((unsigned)((total + 255) / 256)), blk, 0, stream>>>(
        qkv16, img_rope, q16, k16, vt16);
  }

  // 6) flash attention: 8 waves/block share one (b,h); exact grid
  {
    const int ntasks = BATCH * NH * (SEQ / 16);   // 3456
    ssb_attn_kernel<<<dim3(ntasks / 8), blk, 0, stream>>>(q16, k16, vt16, attnO);
  }

  // 7) output projection -> fp32
  ssb_gemm_wmma_kernel<0><<<dim3(DMODEL / 64, NT / 128), blk, 0, stream>>>(
      attnO, wout16, nullptr, c32, NT, DMODEL, DMODEL);

  // 8) x += gate * proj ; RMSNorm2 -> xn16 (reused)
  ssb_resid_norm2_kernel<<<dim3(NT), blk, 0, stream>>>(x, c32, modout, w_norm2, xn16);

  // 9) fc1 + bias + tanh-GELU -> h16
  ssb_gemm_wmma_kernel<2><<<dim3(MLPH / 64, NT / 128), blk, 0, stream>>>(
      xn16, wfc116, b_fc1, h16, NT, MLPH, DMODEL);

  // 10) fc2 -> fp32
  ssb_gemm_wmma_kernel<0><<<dim3(DMODEL / 64, NT / 128), blk, 0, stream>>>(
      h16, wfc216, nullptr, c32, NT, DMODEL, MLPH);

  // 11) final residual + bias, split into (txt_out, img_out)
  {
    const size_t total = (size_t)NT * DMODEL;
    ssb_final_kernel<<<dim3((unsigned)((total + 255) / 256)), blk, 0, stream>>>(
        x, c32, b_fc2, out);
  }
}
